// ScaledDotProduct_65051574665283
// MI455X (gfx1250) — compile-verified, matrix operand order checked by reading, not run on hardware
//
#include <hip/hip_runtime.h>
#include <hip/hip_bf16.h>

typedef __attribute__((ext_vector_type(16))) _Float16 v16h;
typedef __attribute__((ext_vector_type(8)))  float    v8f;

#define BS_   4
#define H_    16
#define SEQ_  2048
#define D_    128
#define KVB   64          // kv rows per block-iteration (2 x WMMA-K for P*V)
#define WAVES 8
#define QROWS (16 * WAVES)
#define KSTR  136         // K tile row stride (halves): 272B, 16B-aligned, conflict-free
#define VTSTR 88          // V^T tile row stride (halves): 176B, 16B-aligned, conflict-free
#define PSTR  88          // P scratch row stride (halves): 176B, 16B-aligned, conflict-free

// xor-butterfly within 16-lane halves via ds_swizzle (SWAPX pattern)
#define SWZ_XOR(x, m) \
  __int_as_float(__builtin_amdgcn_ds_swizzle(__float_as_int(x), ((m) << 10) | 0x1f))

__global__ __launch_bounds__(256, 1)
void ScaledDotProduct_fa_kernel(const float* __restrict__ q,
                                const float* __restrict__ k,
                                const float* __restrict__ v,
                                const float* __restrict__ mask,
                                float* __restrict__ out) {
  __shared__ _Float16 Klds[KVB * KSTR];        // [kv][d]   row-major f16
  __shared__ _Float16 Vt[D_ * VTSTR];          // [d][kv]   transposed f16
  __shared__ float    Mlds[KVB];
  __shared__ _Float16 Plds[WAVES][16 * PSTR];  // per-wave P transpose scratch

  const int tid  = threadIdx.x;
  const int wave = tid >> 5;
  const int lane = tid & 31;
  const int g    = lane >> 4;   // lane half (fragment layouts split at 16)
  const int hn   = lane & 15;   // col (B/C frags) / row (A frags)
  const int bh   = blockIdx.y;
  const int b    = bh / H_;
  const int q0   = blockIdx.x * QROWS + wave * 16;

  const float LOG2E  = 1.4426950408889634f;
  const float qscale = LOG2E * 0.08838834764831845f;  // log2e / sqrt(128)
  const float mscale = -1e9f * LOG2E;

  // ---- Q tile: 16 rows/wave in A-fragment layout, pre-scaled, f32->f16 ----
  // A layout (16-bit 16x32): lane half g, element i -> K index i + 8g + (i&8)
  v16h qa[4];
  {
    const float* qrow = q + ((size_t)bh * SEQ_ + q0 + hn) * D_;
#pragma unroll
    for (int c = 0; c < 4; c++) {
#pragma unroll
      for (int hb = 0; hb < 2; hb++) {
        const int d0 = 32 * c + 16 * hb + 8 * g;
        const float4 f0 = *reinterpret_cast<const float4*>(qrow + d0);
        const float4 f1 = *reinterpret_cast<const float4*>(qrow + d0 + 4);
        const int e = 8 * hb;
        qa[c][e + 0] = (_Float16)(f0.x * qscale);
        qa[c][e + 1] = (_Float16)(f0.y * qscale);
        qa[c][e + 2] = (_Float16)(f0.z * qscale);
        qa[c][e + 3] = (_Float16)(f0.w * qscale);
        qa[c][e + 4] = (_Float16)(f1.x * qscale);
        qa[c][e + 5] = (_Float16)(f1.y * qscale);
        qa[c][e + 6] = (_Float16)(f1.z * qscale);
        qa[c][e + 7] = (_Float16)(f1.w * qscale);
      }
    }
  }

  // all-ones B fragment: P x ones == per-row sum replicated across all 16 cols
  v16h onesb;
#pragma unroll
  for (int i = 0; i < 16; i++) onesb[i] = (_Float16)1.0f;

  v8f acc[8];
#pragma unroll
  for (int t = 0; t < 8; t++) acc[t] = (v8f){0.f,0.f,0.f,0.f,0.f,0.f,0.f,0.f};
  float mrow[8], lrow[8];
#pragma unroll
  for (int r = 0; r < 8; r++) { mrow[r] = -1e30f; lrow[r] = 0.f; }

  const size_t kvbase = (size_t)bh * SEQ_ * D_;

  for (int kv0 = 0; kv0 < SEQ_; kv0 += KVB) {
    // ---- stage K (row-major): 64x128 f32 -> f16 LDS ----
    {
      const float4* Kg = reinterpret_cast<const float4*>(k + kvbase + (size_t)kv0 * D_);
#pragma unroll
      for (int it = 0; it < 8; it++) {
        const int f   = tid + 256 * it;        // float4 index 0..2047
        const int row = f >> 5;                // kv row 0..63
        const int col = (f & 31) << 2;         // d 0..124
        const float4 k4 = Kg[f];
        union { _Float16 h[4]; uint2 u; } pk;
        pk.h[0] = (_Float16)k4.x; pk.h[1] = (_Float16)k4.y;
        pk.h[2] = (_Float16)k4.z; pk.h[3] = (_Float16)k4.w;
        *reinterpret_cast<uint2*>(&Klds[row * KSTR + col]) = pk.u;
      }
    }
    // ---- stage V transposed: 4x4 micro-tile register transpose ----
    {
      const float4* Vg = reinterpret_cast<const float4*>(v + kvbase + (size_t)kv0 * D_);
#pragma unroll
      for (int it = 0; it < 2; it++) {
        const int mt  = tid + 256 * it;        // micro-tile 0..511
        const int kv4 = (mt >> 5) << 2;        // kv base 0,4,..,60
        const int dg  = mt & 31;               // d group
        const int d0  = dg << 2;               // d base 0,4,..,124
        const float4 r0 = Vg[(kv4 + 0) * 32 + dg];
        const float4 r1 = Vg[(kv4 + 1) * 32 + dg];
        const float4 r2 = Vg[(kv4 + 2) * 32 + dg];
        const float4 r3 = Vg[(kv4 + 3) * 32 + dg];
        union { _Float16 h[4]; uint2 u; } c0, c1, c2, c3;
        c0.h[0]=(_Float16)r0.x; c0.h[1]=(_Float16)r1.x; c0.h[2]=(_Float16)r2.x; c0.h[3]=(_Float16)r3.x;
        c1.h[0]=(_Float16)r0.y; c1.h[1]=(_Float16)r1.y; c1.h[2]=(_Float16)r2.y; c1.h[3]=(_Float16)r3.y;
        c2.h[0]=(_Float16)r0.z; c2.h[1]=(_Float16)r1.z; c2.h[2]=(_Float16)r2.z; c2.h[3]=(_Float16)r3.z;
        c3.h[0]=(_Float16)r0.w; c3.h[1]=(_Float16)r1.w; c3.h[2]=(_Float16)r2.w; c3.h[3]=(_Float16)r3.w;
        *reinterpret_cast<uint2*>(&Vt[(d0 + 0) * VTSTR + kv4]) = c0.u;
        *reinterpret_cast<uint2*>(&Vt[(d0 + 1) * VTSTR + kv4]) = c1.u;
        *reinterpret_cast<uint2*>(&Vt[(d0 + 2) * VTSTR + kv4]) = c2.u;
        *reinterpret_cast<uint2*>(&Vt[(d0 + 3) * VTSTR + kv4]) = c3.u;
      }
    }
    if (tid < KVB) Mlds[tid] = mask[(size_t)b * SEQ_ + kv0 + tid];
    __syncthreads();

    // prefetch next kv block (global_prefetch_b8)
    if (kv0 + KVB < SEQ_) {
      const char* pfk = reinterpret_cast<const char*>(k + kvbase + (size_t)(kv0 + KVB) * D_) + tid * 128;
      const char* pfv = reinterpret_cast<const char*>(v + kvbase + (size_t)(kv0 + KVB) * D_) + tid * 128;
      __builtin_prefetch(pfk, 0, 1);
      __builtin_prefetch(pfv, 0, 1);
    }

    // ---- S = Q * K^T : 4 col-tiles x 4 reduction chunks = 16 WMMAs ----
    v8f s[4];
#pragma unroll
    for (int h4 = 0; h4 < 4; h4++) s[h4] = (v8f){0.f,0.f,0.f,0.f,0.f,0.f,0.f,0.f};
#pragma unroll
    for (int c = 0; c < 4; c++) {
#pragma unroll
      for (int h4 = 0; h4 < 4; h4++) {
        v16h kb;
#pragma unroll
        for (int i = 0; i < 16; i++) {
          const int kk = 32 * c + i + (g << 3) + (i & 8);        // d index
          kb[i] = Klds[(16 * h4 + hn) * KSTR + kk];              // kv row
        }
        s[h4] = __builtin_amdgcn_wmma_f32_16x16x32_f16(false, qa[c], false, kb,
                                                       (short)0, s[h4], false, false);
      }
    }

    // additive mask (log2 domain)
#pragma unroll
    for (int h4 = 0; h4 < 4; h4++) {
      const float madd = Mlds[16 * h4 + hn] * mscale;
#pragma unroll
      for (int r = 0; r < 8; r++) s[h4][r] += madd;
    }

    // ---- online softmax: max via ds_swizzle butterfly (row = r + 8g) ----
    float rsc[8];
#pragma unroll
    for (int r = 0; r < 8; r++) {
      float mx = fmaxf(fmaxf(s[0][r], s[1][r]), fmaxf(s[2][r], s[3][r]));
      mx = fmaxf(mx, SWZ_XOR(mx, 1));
      mx = fmaxf(mx, SWZ_XOR(mx, 2));
      mx = fmaxf(mx, SWZ_XOR(mx, 4));
      mx = fmaxf(mx, SWZ_XOR(mx, 8));
      const float mnew = fmaxf(mrow[r], mx);
      rsc[r] = __builtin_amdgcn_exp2f(mrow[r] - mnew);
      mrow[r] = mnew;
      s[0][r] = __builtin_amdgcn_exp2f(s[0][r] - mnew);
      s[1][r] = __builtin_amdgcn_exp2f(s[1][r] - mnew);
      s[2][r] = __builtin_amdgcn_exp2f(s[2][r] - mnew);
      s[3][r] = __builtin_amdgcn_exp2f(s[3][r] - mnew);
    }
#pragma unroll
    for (int t = 0; t < 8; t++)
#pragma unroll
      for (int r = 0; r < 8; r++) acc[t][r] *= rsc[r];

    // ---- transpose P through per-wave LDS: C-layout -> A-fragments ----
    _Float16* P = Plds[wave];
#pragma unroll
    for (int r = 0; r < 8; r++) {
      const int row = r + 8 * g;
      P[row * PSTR + hn]      = (_Float16)s[0][r];
      P[row * PSTR + 16 + hn] = (_Float16)s[1][r];
      P[row * PSTR + 32 + hn] = (_Float16)s[2][r];
      P[row * PSTR + 48 + hn] = (_Float16)s[3][r];
    }
    asm volatile("s_wait_dscnt 0" ::: "memory");
    v16h pa0, pa1;
#pragma unroll
    for (int i = 0; i < 16; i++) {
      const int kk = i + (g << 3) + (i & 8);                     // kv-local
      pa0[i] = P[hn * PSTR + kk];
      pa1[i] = P[hn * PSTR + 32 + kk];
    }

    // ---- row-sum via WMMA with all-ones B (replaces sum butterfly) ----
    v8f rs = (v8f){0.f,0.f,0.f,0.f,0.f,0.f,0.f,0.f};
    rs = __builtin_amdgcn_wmma_f32_16x16x32_f16(false, pa0, false, onesb,
                                                (short)0, rs, false, false);
    rs = __builtin_amdgcn_wmma_f32_16x16x32_f16(false, pa1, false, onesb,
                                                (short)0, rs, false, false);
#pragma unroll
    for (int r = 0; r < 8; r++) lrow[r] = lrow[r] * rsc[r] + rs[r];

    // ---- O += P * V : 8 D-tiles x 2 K-chunks = 16 WMMAs (b128 frag loads) ----
#pragma unroll
    for (int t = 0; t < 8; t++) {
      v16h vb0, vb1;
#pragma unroll
      for (int i = 0; i < 16; i++) {
        const int kk = i + (g << 3) + (i & 8);                   // kv-local
        vb0[i] = Vt[(16 * t + hn) * VTSTR + kk];
        vb1[i] = Vt[(16 * t + hn) * VTSTR + 32 + kk];
      }
      acc[t] = __builtin_amdgcn_wmma_f32_16x16x32_f16(false, pa0, false, vb0,
                                                      (short)0, acc[t], false, false);
      acc[t] = __builtin_amdgcn_wmma_f32_16x16x32_f16(false, pa1, false, vb1,
                                                      (short)0, acc[t], false, false);
    }
    __syncthreads();
  }

  // ---- epilogue: normalize and store ----
  float* orow = out + ((size_t)bh * SEQ_ + q0) * D_;
#pragma unroll
  for (int r = 0; r < 8; r++) {
    const float inv = 1.0f / lrow[r];
    const int row = r + 8 * g;
#pragma unroll
    for (int t = 0; t < 8; t++)
      orow[(size_t)row * D_ + 16 * t + hn] = acc[t][r] * inv;
  }
}

extern "C" void kernel_launch(void* const* d_in, const int* in_sizes, int n_in,
                              void* d_out, int out_size, void* d_ws, size_t ws_size,
                              hipStream_t stream) {
  const float* q    = (const float*)d_in[0];
  const float* k    = (const float*)d_in[1];
  const float* v    = (const float*)d_in[2];
  const float* mask = (const float*)d_in[3];
  float* out = (float*)d_out;
  (void)in_sizes; (void)n_in; (void)out_size; (void)d_ws; (void)ws_size;

  dim3 grid(SEQ_ / QROWS, BS_ * H_);   // (16, 64)
  dim3 block(32 * WAVES);              // 256 threads = 8 wave32
  hipLaunchKernelGGL(ScaledDotProduct_fa_kernel, grid, block, 0, stream,
                     q, k, v, mask, out);
}
